// Joint_Attention_Module_58591943852546
// MI455X (gfx1250) — matrix operand order repaired and checked by reading, not compiled, not run on hardware
//
#include <hip/hip_runtime.h>

typedef __attribute__((ext_vector_type(16))) __bf16 v16bf;
typedef __attribute__((ext_vector_type(8)))  float  v8f;

#define B_   8
#define L_   256
#define C_   512
#define OUT_ 512
#define IT   4   // i-rows per block in the tanh contraction

__device__ __forceinline__ float relu_f(float x) { return x > 0.f ? x : 0.f; }
__device__ __forceinline__ float fast_rcp(float x) { return __builtin_amdgcn_rcpf(x); }

__device__ __forceinline__ float fast_tanh(float x) {
#if __has_builtin(__builtin_amdgcn_tanhf)
    return __builtin_amdgcn_tanhf(x);          // gfx1250 v_tanh_f32
#else
    float e = __expf(2.0f * x);
    return 1.0f - 2.0f * fast_rcp(e + 1.0f);
#endif
}

__device__ __forceinline__ float fast_sigmoid(float x) {
    return fast_rcp(1.0f + __expf(-x));
}

// f32 -> bf16 hi/lo split (error-compensated emulation of f32 MACs on the bf16 WMMA pipe)
struct bf2 { __bf16 h, l; };
__device__ __forceinline__ bf2 bf16_split(float x) {
    bf2 r;
    r.h = (__bf16)x;
    r.l = (__bf16)(x - (float)r.h);
    return r;
}

// -------------------- zero workspace accumulators (runs every call) --------------------
__global__ void init_ws_kernel(float* __restrict__ ssum, float* __restrict__ cp) {
    int t = blockIdx.x * blockDim.x + threadIdx.x;
    if (t < B_) ssum[t] = 0.f;
    if (t < B_ * C_) cp[t] = 0.f;
}

// -------------------- stage 1: pc = relu(xd)@Wc+bc ; pt = relu(xt)@Wp+bp ---------------
// One wave per 16x16 f32 output tile. bf16x3 split GEMM on V_WMMA_F32_16X16X32_BF16:
//   D += Ahi*Bhi (acc_main) ; D += Ahi*Blo + Alo*Bhi (acc_corr)
// A (16x32 bf16) per-lane: K in [k+8*hi, +8) and [k+16+8*hi, +8), element = K offset.
// B (32x16 bf16) per-lane: col n0+l15, K in [k+8*hi, +8) and [k+16+8*hi, +8), element = K offset.
__global__ void proj_kernel(const float* __restrict__ xd, const float* __restrict__ xt,
                            const float* __restrict__ Wc, const float* __restrict__ bc,
                            const float* __restrict__ Wp, const float* __restrict__ bp,
                            float* __restrict__ pc, float* __restrict__ pt) {
    const int lane = threadIdx.x & 31;
    const int l15  = lane & 15;
    const int hi   = lane >> 4;

    int tile = blockIdx.x * (blockDim.x >> 5) + (threadIdx.x >> 5);
    const int tilesPerMat = B_ * (L_ / 16) * (C_ / 16);    // 4096
    const int mat = tile / tilesPerMat;
    tile %= tilesPerMat;
    const int b  = tile / ((L_ / 16) * (C_ / 16));
    tile %= (L_ / 16) * (C_ / 16);
    const int m0 = (tile / (C_ / 16)) * 16;
    const int n0 = (tile % (C_ / 16)) * 16;

    const float* x    = mat ? xt : xd;
    const float* W    = mat ? Wp : Wc;
    const float* bias = mat ? bp : bc;
    float*       out  = mat ? pt : pc;

    const float* xrow = x + (size_t)(b * L_ + m0 + l15) * C_;   // A: row m0+l15
    const float* wcol = W + n0 + l15;                            // B: col n0+l15

    v8f acc_main = {};
    v8f acc_corr = {};
    for (int k = 0; k < C_; k += 32) {
        v16bf a_hi, a_lo, b_hi, b_lo;
        const float* p0 = xrow + k + 8 * hi;        // K chunk 0 (8 contiguous)
        const float* p1 = p0 + 16;                  // K chunk 1 (8 contiguous)
#pragma unroll
        for (int e = 0; e < 8; ++e) {
            bf2 s0 = bf16_split(relu_f(p0[e]));
            bf2 s1 = bf16_split(relu_f(p1[e]));
            a_hi[e]     = s0.h;  a_lo[e]     = s0.l;
            a_hi[e + 8] = s1.h;  a_lo[e + 8] = s1.l;
        }
        const float* wv = wcol + (size_t)(k + 8 * hi) * C_;
#pragma unroll
        for (int e = 0; e < 16; ++e) {
            bf2 s = bf16_split(wv[(size_t)(e < 8 ? e : e + 8) * C_]);
            b_hi[e] = s.h;  b_lo[e] = s.l;
        }

        acc_main = __builtin_amdgcn_wmma_f32_16x16x32_bf16(false, a_hi, false, b_hi,
                                                           (short)0, acc_main, false, false);
        acc_corr = __builtin_amdgcn_wmma_f32_16x16x32_bf16(false, a_hi, false, b_lo,
                                                           (short)0, acc_corr, false, false);
        acc_corr = __builtin_amdgcn_wmma_f32_16x16x32_bf16(false, a_lo, false, b_hi,
                                                           (short)0, acc_corr, false, false);
    }

    const float bb    = bias[n0 + l15];
    const int   rbase = m0 + (hi << 3);
    float* orow = out + (size_t)b * L_ * C_ + n0 + l15;
#pragma unroll
    for (int r = 0; r < 8; ++r)
        orow[(size_t)(rbase + r) * C_] = acc_main[r] + acc_corr[r] + bb;
}

// -------------------- stage 2: scores = sigmoid(pt . pc^T), ssum[b] = sum(scores[b]) ----
// Same bf16x3 scheme; both operands contract over c, so B loads are contiguous per lane.
__global__ void score_kernel(const float* __restrict__ pt, const float* __restrict__ pc,
                             float* __restrict__ scores, float* __restrict__ ssum) {
    const int lane = threadIdx.x & 31;
    const int l15  = lane & 15;
    const int hi   = lane >> 4;

    int tile = blockIdx.x * (blockDim.x >> 5) + (threadIdx.x >> 5);
    const int b  = tile / ((L_ / 16) * (L_ / 16));
    tile %= (L_ / 16) * (L_ / 16);
    const int i0 = (tile / (L_ / 16)) * 16;
    const int j0 = (tile % (L_ / 16)) * 16;

    const float* arow = pt + (size_t)(b * L_ + i0 + l15) * C_;   // A: pt row i0+l15
    const float* brow = pc + (size_t)(b * L_ + j0 + l15) * C_;   // B: pc row j0+l15 (col of B)

    v8f acc_main = {};
    v8f acc_corr = {};
    for (int k = 0; k < C_; k += 32) {
        v16bf a_hi, a_lo, b_hi, b_lo;
        const float* p0 = arow + k + 8 * hi;
        const float* p1 = p0 + 16;
#pragma unroll
        for (int e = 0; e < 8; ++e) {
            bf2 s0 = bf16_split(p0[e]);
            bf2 s1 = bf16_split(p1[e]);
            a_hi[e]     = s0.h;  a_lo[e]     = s0.l;
            a_hi[e + 8] = s1.h;  a_lo[e + 8] = s1.l;
        }
        const float* q = brow + k + 8 * hi;
#pragma unroll
        for (int e = 0; e < 16; ++e) {
            bf2 s = bf16_split(q[e < 8 ? e : e + 8]);
            b_hi[e] = s.h;  b_lo[e] = s.l;
        }

        acc_main = __builtin_amdgcn_wmma_f32_16x16x32_bf16(false, a_hi, false, b_hi,
                                                           (short)0, acc_main, false, false);
        acc_corr = __builtin_amdgcn_wmma_f32_16x16x32_bf16(false, a_hi, false, b_lo,
                                                           (short)0, acc_corr, false, false);
        acc_corr = __builtin_amdgcn_wmma_f32_16x16x32_bf16(false, a_lo, false, b_hi,
                                                           (short)0, acc_corr, false, false);
    }

    float psum = 0.f;
    const int rbase = i0 + (hi << 3);
    float* srow = scores + (size_t)b * L_ * L_ + j0 + l15;
#pragma unroll
    for (int r = 0; r < 8; ++r) {
        float s = fast_sigmoid(acc_main[r] + acc_corr[r]);
        srow[(size_t)(rbase + r) * L_] = s;
        psum += s;
    }
    // wave32 reduction, then one atomic per wave
#pragma unroll
    for (int off = 16; off > 0; off >>= 1)
        psum += __shfl_xor(psum, off, 32);
    if (lane == 0) atomicAdd(&ssum[b], psum);
}

// -------------------- stage 3: cp[b,c] = (1/S[b]) * sum_ij tanh(xd*xt) * w -------------
// j outer / i inner: each xt vector load pair is reused across IT rows of i,
// giving 2*IT tanh+FMA per 2 vector loads + IT uniform scalar loads.
__global__ void tanh_contract_kernel(const float* __restrict__ xd, const float* __restrict__ xt,
                                     const float* __restrict__ scores,
                                     const float* __restrict__ ssum, float* __restrict__ cp) {
    const int b  = blockIdx.x / (L_ / IT);
    const int i0 = (blockIdx.x % (L_ / IT)) * IT;
    const int c  = threadIdx.x;                 // handles c and c+256

    const float inv = fast_rcp(ssum[b]);

    const float* xtb = xt + (size_t)b * L_ * C_;
    const float* wb  = scores + (size_t)b * L_ * L_ + (size_t)i0 * L_;

    float xd0[IT], xd1[IT];
#pragma unroll
    for (int ii = 0; ii < IT; ++ii) {
        const float* xr = xd + (size_t)(b * L_ + i0 + ii) * C_;
        xd0[ii] = xr[c];
        xd1[ii] = xr[c + 256];
    }

    float acc0 = 0.f, acc1 = 0.f;
    for (int j = 0; j < L_; ++j) {
        const float t0 = xtb[(size_t)j * C_ + c];
        const float t1 = xtb[(size_t)j * C_ + c + 256];
#pragma unroll
        for (int ii = 0; ii < IT; ++ii) {
            const float wj = wb[(size_t)ii * L_ + j];   // uniform -> s_load
            acc0 += fast_tanh(xd0[ii] * t0) * wj;
            acc1 += fast_tanh(xd1[ii] * t1) * wj;
        }
    }
    atomicAdd(&cp[(size_t)b * C_ + c],       acc0 * inv);
    atomicAdd(&cp[(size_t)b * C_ + c + 256], acc1 * inv);
}

// -------------------- stage 4: out = cp @ Wf + bf (8x512x512, negligible) --------------
__global__ void final_kernel(const float* __restrict__ cp, const float* __restrict__ Wf,
                             const float* __restrict__ bf, float* __restrict__ out) {
    const int b = blockIdx.x;
    const int o = threadIdx.x;
    float acc0 = bf[o], acc1 = bf[o + 256];
    const float* cpb = cp + (size_t)b * C_;
#pragma unroll 4
    for (int k = 0; k < C_; ++k) {
        const float cv = cpb[k];                       // uniform -> s_load
        acc0 += cv * Wf[(size_t)k * OUT_ + o];
        acc1 += cv * Wf[(size_t)k * OUT_ + o + 256];
    }
    out[(size_t)b * OUT_ + o]       = acc0;
    out[(size_t)b * OUT_ + o + 256] = acc1;
}

extern "C" void kernel_launch(void* const* d_in, const int* in_sizes, int n_in,
                              void* d_out, int out_size, void* d_ws, size_t ws_size,
                              hipStream_t stream) {
    const float* xd = (const float*)d_in[0];
    const float* xt = (const float*)d_in[1];
    const float* Wc = (const float*)d_in[2];
    const float* bc = (const float*)d_in[3];
    const float* Wp = (const float*)d_in[4];
    const float* bp = (const float*)d_in[5];
    const float* Wf = (const float*)d_in[6];
    const float* bf = (const float*)d_in[7];
    float* out = (float*)d_out;

    float* ws     = (float*)d_ws;
    float* pt     = ws;                                  // B*L*C
    float* pc     = pt + (size_t)B_ * L_ * C_;           // B*L*C
    float* scores = pc + (size_t)B_ * L_ * C_;           // B*L*L
    float* ssum   = scores + (size_t)B_ * L_ * L_;       // B
    float* cp     = ssum + B_;                           // B*C

    init_ws_kernel<<<(B_ * C_ + 255) / 256, 256, 0, stream>>>(ssum, cp);

    // 2 matrices * 8 b * 16 mtiles * 32 ntiles = 8192 waves; 4 waves/block
    proj_kernel<<<(2 * B_ * (L_ / 16) * (C_ / 16)) / 4, 128, 0, stream>>>(
        xd, xt, Wc, bc, Wp, bp, pc, pt);

    // 8 b * 16 * 16 = 2048 waves; 4 waves/block
    score_kernel<<<(B_ * (L_ / 16) * (L_ / 16)) / 4, 128, 0, stream>>>(
        pt, pc, scores, ssum);

    tanh_contract_kernel<<<B_ * (L_ / IT), 256, 0, stream>>>(xd, xt, scores, ssum, cp);

    final_kernel<<<B_, 256, 0, stream>>>(cp, Wf, bf, out);
}
